// TorchAttractorLanguageModel_87565793231038
// MI455X (gfx1250) — 8-bit matrix operand order repaired and checked by reading, compiled, not run on hardware
//
#include <hip/hip_runtime.h>
#include <stdint.h>


typedef __attribute__((ext_vector_type(16))) int    v16i;
typedef __attribute__((ext_vector_type(8)))  float  v8f;
typedef __attribute__((ext_vector_type(16))) __bf16 v16bf;

union FragI { v16i  v; uint32_t u[16]; };
union FragB { v16bf v; uint32_t u[8];  };

__device__ __forceinline__ v8f zero8() {
  v8f z;
#pragma unroll
  for (int i = 0; i < 8; ++i) z[i] = 0.0f;
  return z;
}

__device__ __forceinline__ uint32_t f32_to_e4m3(float x) {
  uint32_t sgn = (__float_as_uint(x) >> 31) << 7;
  float a = fminf(fabsf(x), 448.0f);
  uint32_t au   = __float_as_uint(a);
  uint32_t norm = ((au + 0x80000u) >> 20) - 960u;
  uint32_t den  = (uint32_t)(int)(a * 512.0f + 0.5f);
  uint32_t mag  = (a < 0.015625f) ? den : norm;
  return sgn | mag;
}

__device__ __forceinline__ uint32_t pack4_e4m3(float a, float b, float c, float d) {
  return f32_to_e4m3(a) | (f32_to_e4m3(b) << 8) |
         (f32_to_e4m3(c) << 16) | (f32_to_e4m3(d) << 24);
}

__device__ __forceinline__ uint32_t f32_to_bf16(float f) {
  uint32_t u = __float_as_uint(f);
  u += 0x7FFFu + ((u >> 16) & 1u);
  return u >> 16;
}
__device__ __forceinline__ uint32_t pack2_bf16(float a, float b) {
  return f32_to_bf16(a) | (f32_to_bf16(b) << 16);
}

__device__ __forceinline__ float tanh_fast(float x) {
#if __has_builtin(__builtin_amdgcn_tanh_f32)
  return __builtin_amdgcn_tanh_f32(x);
#elif __has_builtin(__builtin_amdgcn_tanhf)
  return __builtin_amdgcn_tanhf(x);
#else
  float e = __builtin_amdgcn_exp2f(x * 2.8853900817779268f);
  return 1.0f - 2.0f * __builtin_amdgcn_rcpf(e + 1.0f);
#endif
}

__device__ __forceinline__ int a8_kbase(int j, int hi) {
  return ((j >> 3) << 6) + (((j & 7) >> 1) << 4) + ((j & 1) << 2) + (hi << 3);
}
__device__ __forceinline__ int b8_kbase(int j, int hi) {
  return ((j >> 3) << 6) + (((j & 7) >> 1) << 4) + ((j & 1) << 2) + (hi << 3);
}
__device__ __forceinline__ int bf_kpair(int v, int hi) {
  return ((v >> 2) << 4) + ((v & 3) << 1) + (hi << 3);
}

__device__ __forceinline__ v8f wmma_fp8(v16i a, v16i b, v8f c) {
  return __builtin_amdgcn_wmma_f32_16x16x128_fp8_fp8(a, b, (short)0, c, false, false);
}
__device__ __forceinline__ v8f wmma_bf16w(v16bf a, v16bf b, v8f c) {
  return __builtin_amdgcn_wmma_f32_16x16x32_bf16(false, a, false, b, (short)0, c,
                                                 false, false);
}

#define VOC   32000
#define DD    512
#define HR    256
#define LL    512
#define DTC   0.05f
#define CUBIC 0.05f
#define CPLH  0.0025f

__global__ __launch_bounds__(32) void k_prep_w1(const float* __restrict__ Vm,
                                                uint8_t* __restrict__ W1f) {
  int f  = blockIdx.x;
  int kt = f >> 4, nt = f & 15;
  int lane = threadIdx.x, lo = lane & 15, hi = lane >> 4;
  int hr = nt * 16 + lo, h = hr >> 6, r = hr & 63;
  const float* row = Vm + (size_t)(h * 64 + r) * DD;
  uint32_t* dst = (uint32_t*)(W1f + (size_t)f * 2048 + lane * 64);
#pragma unroll
  for (int j = 0; j < 16; ++j) {
    int kb = kt * 128 + b8_kbase(j, hi);
    float4 w = *(const float4*)&row[kb];
    dst[j] = pack4_e4m3(w.x, w.y, w.z, w.w);
  }
}

__global__ __launch_bounds__(32) void k_prep_w2(const float* __restrict__ U,
                                                uint8_t* __restrict__ W2f) {
  int f  = blockIdx.x;
  int kt = f >> 5, nt = f & 31;
  int lane = threadIdx.x, lo = lane & 15, hi = lane >> 4;
  int d = nt * 16 + lo;
  uint32_t* dst = (uint32_t*)(W2f + (size_t)f * 2048 + lane * 64);
#pragma unroll
  for (int j = 0; j < 16; ++j) {
    int kb = kt * 128 + b8_kbase(j, hi);
    int h = kb >> 6, r0 = kb & 63;
    float4 w = *(const float4*)&U[((size_t)(h * DD) + d) * 64 + r0];
    dst[j] = pack4_e4m3(w.x, w.y, w.z, w.w);
  }
}

__global__ __launch_bounds__(256) void k_attractor(const float* __restrict__ embed,
                                                   const uint8_t* __restrict__ W1f,
                                                   const uint8_t* __restrict__ W2f,
                                                   float* __restrict__ attr) {
  extern __shared__ uint8_t smem[];
  float*   Sl = (float*)smem;
  uint8_t* Zl = smem + 64 * DD * 4;

  const int tid = threadIdx.x, lane = tid & 31, wid = tid >> 5;
  const int lo = lane & 15, hi = lane >> 4;
  const int row0 = blockIdx.x * 64;
  const int mt = wid & 3, ng = wid >> 2;

  for (int i = tid; i < 64 * DD; i += 256) Sl[i] = 0.0f;
  __syncthreads();

#pragma unroll 1
  for (int step = 0; step < 16; ++step) {
    FragI afr[4];
#pragma unroll
    for (int kt = 0; kt < 4; ++kt) {
#pragma unroll
      for (int j = 0; j < 16; ++j) {
        int kb = kt * 128 + a8_kbase(j, hi);
        float4 s4 = *(const float4*)&Sl[(mt * 16 + lo) * DD + kb];
        afr[kt].u[j] = pack4_e4m3(s4.x, s4.y, s4.z, s4.w);
      }
    }
#pragma unroll 1
    for (int i = 0; i < 8; ++i) {
      int nt = ng * 8 + i;
      v8f acc = zero8();
#pragma unroll
      for (int kt = 0; kt < 4; ++kt) {
        FragI bfr;
        bfr.v = *(const v16i*)(W1f + (size_t)(kt * 16 + nt) * 2048 + lane * 64);
        acc = wmma_fp8(afr[kt].v, bfr.v, acc);
      }
#pragma unroll
      for (int e = 0; e < 8; ++e) {
        int m = mt * 16 + e + hi * 8;
        int n = nt * 16 + lo;
        Zl[m * HR + n] = (uint8_t)f32_to_e4m3(tanh_fast(acc[e]));
      }
    }
    __syncthreads();

    FragI zfr[2];
#pragma unroll
    for (int kt = 0; kt < 2; ++kt) {
#pragma unroll
      for (int j = 0; j < 16; ++j) {
        int kb = kt * 128 + a8_kbase(j, hi);
        zfr[kt].u[j] = *(const uint32_t*)&Zl[(mt * 16 + lo) * HR + kb];
      }
    }
#pragma unroll 1
    for (int i = 0; i < 16; ++i) {
      int nt = ng * 16 + i;
      v8f acc = zero8();
#pragma unroll
      for (int kt = 0; kt < 2; ++kt) {
        FragI bfr;
        bfr.v = *(const v16i*)(W2f + (size_t)(kt * 32 + nt) * 2048 + lane * 64);
        acc = wmma_fp8(zfr[kt].v, bfr.v, acc);
      }
#pragma unroll
      for (int e = 0; e < 8; ++e) {
        int m = mt * 16 + e + hi * 8;
        int n = nt * 16 + lo;
        float s  = Sl[m * DD + n];
        float sg = embed[(size_t)(row0 + m) * DD + n];
        float fld = sg - s - CUBIC * s * s * s + CPLH * acc[e];
        Sl[m * DD + n] = s + DTC * fld;
      }
    }
    __syncthreads();
  }

  for (int i = tid; i < 64 * DD; i += 256)
    attr[(size_t)row0 * DD + i] = Sl[i];
}

__global__ __launch_bounds__(32) void k_swz_attr(const float* __restrict__ attr,
                                                 uint32_t* __restrict__ aswz) {
  int f  = blockIdx.x;
  int kt = f / 2000, nt = f % 2000;
  int lane = threadIdx.x, lo = lane & 15, hi = lane >> 4;
  const float* row = attr + (size_t)(nt * 16 + lo) * DD;
  uint32_t* dst = aswz + (size_t)f * 256 + lane * 8;
#pragma unroll
  for (int v = 0; v < 8; ++v) {
    int kb = kt * 32 + bf_kpair(v, hi);
    float2 w = *(const float2*)&row[kb];
    dst[v] = pack2_bf16(w.x, w.y);
  }
}

__global__ __launch_bounds__(32) void k_norms(const float* __restrict__ rows,
                                              float* __restrict__ out) {
  const float* p = rows + (size_t)blockIdx.x * DD;
  float s = 0.0f;
  for (int c = threadIdx.x; c < DD; c += 32) { float v = p[c]; s += v * v; }
#pragma unroll
  for (int off = 16; off > 0; off >>= 1) s += __shfl_xor(s, off, 32);
  if (threadIdx.x == 0) out[blockIdx.x] = s;
}

__global__ __launch_bounds__(1024) void k_converge(const int* __restrict__ ids,
                                                   const float* __restrict__ embed,
                                                   const uint8_t* __restrict__ W1f,
                                                   const uint8_t* __restrict__ W2f,
                                                   float* __restrict__ sconv) {
  extern __shared__ uint8_t smem[];
  uint8_t* W1L  = smem;
  uint8_t* W2L  = smem + 131072;
  float*   sF   = (float*)(smem + 262144);
  float*   sigF = (float*)(smem + 264192);
  uint8_t* s8   = smem + 266240;
  uint8_t* z8   = smem + 266752;

  const int tid = threadIdx.x, lane = tid & 31, wid = tid >> 5;
  const int lo = lane & 15, hi = lane >> 4;

  {
    const uint4* s1 = (const uint4*)W1f; uint4* d1 = (uint4*)W1L;
    const uint4* s2 = (const uint4*)W2f; uint4* d2 = (uint4*)W2L;
    for (int i = tid; i < 8192; i += 1024) { d1[i] = s1[i]; d2[i] = s2[i]; }
  }
  if (tid < DD) sF[tid] = 0.0f;
  __syncthreads();

#pragma unroll 1
  for (int t = 0; t < LL; ++t) {
    int tok = ids[t];
    if (tid < DD) sigF[tid] = embed[(size_t)tok * DD + tid];
    __syncthreads();

#pragma unroll 1
    for (int step = 0; step < 12; ++step) {
      if (tid < DD) s8[tid] = (uint8_t)f32_to_e4m3(sF[tid]);
      __syncthreads();

      if (wid < 16) {
        v8f acc = zero8();
#pragma unroll
        for (int kt = 0; kt < 4; ++kt) {
          FragI a;
#pragma unroll
          for (int j = 0; j < 16; ++j) {
            int kb = kt * 128 + a8_kbase(j, hi);
            a.u[j] = (lo == 0) ? *(const uint32_t*)&s8[kb] : 0u;
          }
          FragI b;
          b.v = *(const v16i*)(W1L + (size_t)(kt * 16 + wid) * 2048 + lane * 64);
          acc = wmma_fp8(a.v, b.v, acc);
        }
        if (lane < 16)
          z8[wid * 16 + lane] = (uint8_t)f32_to_e4m3(tanh_fast(acc[0]));
      }
      __syncthreads();

      {
        v8f acc = zero8();
#pragma unroll
        for (int kt = 0; kt < 2; ++kt) {
          FragI a;
#pragma unroll
          for (int j = 0; j < 16; ++j) {
            int kb = kt * 128 + a8_kbase(j, hi);
            a.u[j] = (lo == 0) ? *(const uint32_t*)&z8[kb] : 0u;
          }
          FragI b;
          b.v = *(const v16i*)(W2L + (size_t)(kt * 32 + wid) * 2048 + lane * 64);
          acc = wmma_fp8(a.v, b.v, acc);
        }
        if (lane < 16) {
          int n = wid * 16 + lane;
          float s   = sF[n];
          float fld = sigF[n] - s - CUBIC * s * s * s + CPLH * acc[0];
          sF[n] = s + DTC * fld;
        }
      }
      __syncthreads();
    }

    if (tid < DD) sconv[(size_t)t * DD + tid] = sF[tid];
    __syncthreads();
  }
}

__global__ __launch_bounds__(256) void k_dist(const float* __restrict__ sconv,
                                              const uint32_t* __restrict__ aswz,
                                              const float* __restrict__ na,
                                              const float* __restrict__ ns,
                                              const float* __restrict__ temp,
                                              float* __restrict__ out) {
  extern __shared__ uint8_t smem[];
  uint8_t* AF  = smem;
  float*   nsL = (float*)(smem + 65536);
  float*   naL = (float*)(smem + 65536 + 256);

  const int tid = threadIdx.x, lane = tid & 31, wid = tid >> 5;
  const int lo = lane & 15, hi = lane >> 4;
  const int mb = blockIdx.x / 250, nb = blockIdx.x % 250;
  const int row0 = mb * 64, col0 = nb * 128;

#pragma unroll 1
  for (int i = 0; i < 8; ++i) {
    int f  = wid * 8 + i;
    int mt = f >> 4, kt = f & 15;
    const float* row = sconv + (size_t)(row0 + mt * 16 + lo) * DD;
    uint32_t* dst = (uint32_t*)(AF + (size_t)f * 1024) + lane * 8;
#pragma unroll
    for (int v = 0; v < 8; ++v) {
      int kb = kt * 32 + bf_kpair(v, hi);
      float2 w = *(const float2*)&row[kb];
      dst[v] = pack2_bf16(w.x, w.y);
    }
  }
  if (tid < 64)  nsL[tid] = ns[row0 + tid];
  if (tid < 128) naL[tid] = na[col0 + tid];
  __syncthreads();

  float invTau = -1.0f / fmaxf(temp[0], 1e-6f);
  const int mt = wid & 3, ngg = wid >> 2;
#pragma unroll 1
  for (int i = 0; i < 4; ++i) {
    int nt = ngg + 2 * i;
    v8f acc = zero8();
#pragma unroll
    for (int kt = 0; kt < 16; ++kt) {
      FragB a, b;
      a.v = *(const v16bf*)(AF + (size_t)(mt * 16 + kt) * 1024 + lane * 32);
      b.v = *(const v16bf*)((const uint8_t*)aswz +
                            ((size_t)(kt * 2000) + (col0 >> 4) + nt) * 1024 +
                            lane * 32);
      acc = wmma_bf16w(a.v, b.v, acc);
    }
#pragma unroll
    for (int e = 0; e < 8; ++e) {
      int m = mt * 16 + e + hi * 8;
      int n = nt * 16 + lo;
      float d2 = nsL[m] + naL[n] - 2.0f * acc[e];
      out[(size_t)(row0 + m) * VOC + col0 + n] = invTau * sqrtf(fmaxf(d2, 0.0f));
    }
  }
}

extern "C" void kernel_launch(void* const* d_in, const int* in_sizes, int n_in,
                              void* d_out, int out_size, void* d_ws, size_t ws_size,
                              hipStream_t stream) {
  (void)in_sizes; (void)n_in; (void)out_size; (void)ws_size;
  const int*   ids   = (const int*)d_in[0];
  const float* embed = (const float*)d_in[1];
  const float* U     = (const float*)d_in[2];
  const float* Vm    = (const float*)d_in[3];
  const float* temp  = (const float*)d_in[4];
  float*       out   = (float*)d_out;

  uint8_t*  ws    = (uint8_t*)d_ws;
  float*    attr  = (float*)ws;
  uint32_t* aswz  = (uint32_t*)(ws + 65536000);
  uint8_t*  w1f   = ws + 98304000;
  uint8_t*  w2f   = ws + 98435072;
  float*    sconv = (float*)(ws + 98566144);
  float*    na    = (float*)(ws + 99614720);
  float*    nsq   = (float*)(ws + 99742720);

  k_prep_w1  <<<64,    32,   0,      stream>>>(Vm, w1f);
  k_prep_w2  <<<64,    32,   0,      stream>>>(U,  w2f);
  k_attractor<<<500,   256,  147456, stream>>>(embed, w1f, w2f, attr);
  k_swz_attr <<<32000, 32,   0,      stream>>>(attr, aswz);
  k_norms    <<<32000, 32,   0,      stream>>>(attr, na);
  k_converge <<<1,     1024, 267008, stream>>>(ids, embed, w1f, w2f, sconv);
  k_norms    <<<512,   32,   0,      stream>>>(sconv, nsq);
  k_dist     <<<2000,  256,  66304,  stream>>>(sconv, aswz, na, nsq, temp, out);
}
